// Gemma4MoEBlock_26113401160078
// MI455X (gfx1250) — matrix operand order repaired and checked by reading, not compiled, not run on hardware
//
#include <hip/hip_runtime.h>
#include <hip/hip_bf16.h>
#include <math.h>

// ---------------------------------------------------------------------------
// Gemma4 MoE block for gfx1250 (MI455X).
// Weight-streaming-bound: 768MB fp32 expert weights read ~once from HBM
// (M=128 token tiles -> re-read factor ~1), converted in-register to bf16,
// fed to v_wmma_f32_16x16x32_bf16. GEMM2 A-tiles staged with
// global_load_async_to_lds_b128 (ASYNCcnt) + double-buffered LDS.
// ---------------------------------------------------------------------------

typedef __attribute__((ext_vector_type(16))) __bf16 v16bf;
typedef __attribute__((ext_vector_type(8)))  float  v8f;

#define WMMA_BF16(A, B, C) \
  __builtin_amdgcn_wmma_f32_16x16x32_bf16(false, (A), false, (B), (short)0, (C), false, false)

static constexpr int Hdim = 1024;
static constexpr int Edim = 128;
static constexpr int MIDd = 512;
static constexpr int CAPd = 512;
static constexpr int KTOP = 8;

// ---- helpers ---------------------------------------------------------------

__device__ __forceinline__ float gelu_tanh(float v) {
  const float c = 0.7978845608028654f;  // sqrt(2/pi)
  float u = c * (v + 0.044715f * v * v * v);
  return 0.5f * v * (1.0f + tanhf(u));
}

// A-matrix 16x32 bf16 fragment from an LDS tile (row stride 32).
// lane r (0..15) holds row M=r; half=lane>>4 selects K {0..15}/{16..31}.
__device__ __forceinline__ v16bf lds_a_frag(const __bf16* As, int row, int half) {
  v16bf a;
  const __bf16* p0 = As + row * 32 + half * 16;
  const __bf16* p1 = As + row * 32 + half * 16 + 8;
#pragma unroll
  for (int j = 0; j < 8; ++j) { a[j] = p0[j]; a[8 + j] = p1[j]; }
  return a;
}

// B-matrix 32x16 bf16 fragment: lane holds 16 consecutive K of its column
// (weights stored [N][K] row-major => each lane streams 64B of fp32 once).
__device__ __forceinline__ v16bf glb_b_frag_f32(const float* p) {
  float4 f0 = ((const float4*)p)[0];
  float4 f1 = ((const float4*)p)[1];
  float4 f2 = ((const float4*)p)[2];
  float4 f3 = ((const float4*)p)[3];
  v16bf b;
  b[0]  = (__bf16)f0.x; b[1]  = (__bf16)f0.y; b[2]  = (__bf16)f0.z; b[3]  = (__bf16)f0.w;
  b[4]  = (__bf16)f1.x; b[5]  = (__bf16)f1.y; b[6]  = (__bf16)f1.z; b[7]  = (__bf16)f1.w;
  b[8]  = (__bf16)f2.x; b[9]  = (__bf16)f2.y; b[10] = (__bf16)f2.z; b[11] = (__bf16)f2.w;
  b[12] = (__bf16)f3.x; b[13] = (__bf16)f3.y; b[14] = (__bf16)f3.z; b[15] = (__bf16)f3.w;
  return b;
}

union Pack16 { __bf16 h[16]; uint4 q[2]; };

// Async global->LDS DMA, GVS addressing: mem = saddr(64b SGPR) + vaddr(32b byte
// offset). VDST VGPR carries the LDS byte address (low 32 bits of generic ptr).
__device__ __forceinline__ void async_b128(unsigned lds_addr, unsigned goff,
                                           unsigned long long base) {
  asm volatile("global_load_async_to_lds_b128 %0, %1, %2"
               :: "v"(lds_addr), "v"(goff), "s"(base) : "memory");
}
__device__ __forceinline__ void wait_async0() {
  asm volatile("s_wait_asynccnt 0" ::: "memory");
}

// ---- kernel 0: zero output + expert counters -------------------------------

__global__ void zero_init(float* __restrict__ out, int n_out, int* __restrict__ counts) {
  int i = blockIdx.x * blockDim.x + threadIdx.x;
  if (i < n_out) out[i] = 0.0f;
  if (i < Edim) counts[i] = 0;
}

// ---- kernel 1: router (fp32 exact) -----------------------------------------

__global__ __launch_bounds__(128)
void router_topk(const float* __restrict__ x, const float* __restrict__ rscale,
                 const float* __restrict__ rw, const float* __restrict__ pescale,
                 int* __restrict__ topk_idx, float* __restrict__ topk_sc) {
  __shared__ __align__(16) float xs[Hdim];
  __shared__ float sc[Edim];
  __shared__ float red_v[Edim];
  __shared__ int   red_i[Edim];

  const int t = blockIdx.x;
  const int tid = threadIdx.x;

  for (int h = tid; h < Hdim; h += 128) xs[h] = x[(size_t)t * Hdim + h] * rscale[h];
  __syncthreads();

  const float4* wv = (const float4*)(rw + (size_t)tid * Hdim);
  const float4* xv = (const float4*)xs;
  float acc = 0.0f;
#pragma unroll 4
  for (int h = 0; h < Hdim / 4; ++h) {
    float4 a = xv[h], b = wv[h];
    acc += a.x * b.x + a.y * b.y + a.z * b.z + a.w * b.w;
  }
  float logit = acc * pescale[tid];
  sc[tid] = 1.0f / (1.0f + expf(-logit));
  __syncthreads();

  for (int k = 0; k < KTOP; ++k) {
    red_v[tid] = sc[tid];
    red_i[tid] = tid;
    __syncthreads();
#pragma unroll
    for (int s = 64; s > 0; s >>= 1) {
      if (tid < s) {
        float ov = red_v[tid + s]; int oi = red_i[tid + s];
        if (ov > red_v[tid] || (ov == red_v[tid] && oi < red_i[tid])) {
          red_v[tid] = ov; red_i[tid] = oi;
        }
      }
      __syncthreads();
    }
    if (tid == 0) {
      topk_idx[t * KTOP + k] = red_i[0];
      topk_sc [t * KTOP + k] = red_v[0];
      sc[red_i[0]] = -1.0f;  // sigmoid in (0,1): -1 removes it
    }
    __syncthreads();
  }
}

// ---- kernel 2: slot assignment per expert ----------------------------------

__global__ void assign_slots(const int* __restrict__ topk_idx,
                             const float* __restrict__ topk_sc,
                             int* __restrict__ counts,
                             int* __restrict__ etok, float* __restrict__ esc,
                             int n_assign) {
  int a = blockIdx.x * blockDim.x + threadIdx.x;
  if (a >= n_assign) return;
  int e = topk_idx[a];
  int pos = atomicAdd(&counts[e], 1);
  if (pos < CAPd) {
    etok[e * CAPd + pos] = a >> 3;       // token id (K=8)
    esc [e * CAPd + pos] = topk_sc[a];   // sigmoid gate score
  }
}

// ---- kernel 3: GEMM1 + fused gelu*up ---------------------------------------
// Block = 8 waves. Tile: M=128 tokens, 128 "j" columns of MID; each block also
// computes the matching "up" columns (j+MID) so gelu(gate)*up fuses in regs.
// Double-buffered LDS A staging (fp32 gather -> bf16 convert -> ds_store_b128).
// Grid: (MID/128, CAP/128, E).

__global__ __launch_bounds__(256)
void moe_gemm1(const float* __restrict__ x, const float* __restrict__ gup,
               const int* __restrict__ counts, const int* __restrict__ etok,
               __bf16* __restrict__ hbuf) {
  const int e = blockIdx.z;
  int C = counts[e]; if (C > CAPd) C = CAPd;
  const int mbase = blockIdx.y * 128;
  if (mbase >= C) return;
  const int nbase = blockIdx.x * 128;

  const int tid  = threadIdx.x;
  const int wave = tid >> 5, lane = tid & 31;
  const int half = lane >> 4, r = lane & 15;
  const int jcol = nbase + wave * 16 + r;  // 0..MID-1

  __shared__ __align__(16) __bf16 As[2][128 * 32];

  // staging plan: thread -> (row, 16 consecutive k) of the 128x32 tile
  const int arow = tid >> 1;
  const int akk  = (tid & 1) * 16;
  int tok = -1;
  if (mbase + arow < C) tok = etok[e * CAPd + mbase + arow];
  const float* xrow = (tok >= 0) ? (x + (size_t)tok * Hdim + akk) : nullptr;

  const float* wg = gup + ((size_t)e * 2 * MIDd + jcol) * Hdim + half * 16;
  const float* wu = gup + ((size_t)e * 2 * MIDd + (jcol + MIDd)) * Hdim + half * 16;

  auto stage = [&](int b, int k0) {
    float4 f0 = make_float4(0.f, 0.f, 0.f, 0.f), f1 = f0, f2 = f0, f3 = f0;
    if (tok >= 0) {
      const float4* p = (const float4*)(xrow + k0);
      f0 = p[0]; f1 = p[1]; f2 = p[2]; f3 = p[3];
    }
    Pack16 u;
    u.h[0]  = (__bf16)f0.x; u.h[1]  = (__bf16)f0.y; u.h[2]  = (__bf16)f0.z; u.h[3]  = (__bf16)f0.w;
    u.h[4]  = (__bf16)f1.x; u.h[5]  = (__bf16)f1.y; u.h[6]  = (__bf16)f1.z; u.h[7]  = (__bf16)f1.w;
    u.h[8]  = (__bf16)f2.x; u.h[9]  = (__bf16)f2.y; u.h[10] = (__bf16)f2.z; u.h[11] = (__bf16)f2.w;
    u.h[12] = (__bf16)f3.x; u.h[13] = (__bf16)f3.y; u.h[14] = (__bf16)f3.z; u.h[15] = (__bf16)f3.w;
    uint4* d = (uint4*)&As[b][arow * 32 + akk];
    d[0] = u.q[0]; d[1] = u.q[1];
  };

  v8f accg[8] = {};
  v8f accu[8] = {};

  constexpr int NSTEP = Hdim / 32;  // 32 k-steps
  stage(0, 0);
  __syncthreads();

  for (int ks = 0; ks < NSTEP; ++ks) {
    const int cur = ks & 1;
    const int k0  = ks * 32;
    if (ks + 1 < NSTEP) {
      __builtin_prefetch(wg + k0 + 32, 0, 0);  // global_prefetch_b8
      __builtin_prefetch(wu + k0 + 32, 0, 0);
      stage(cur ^ 1, k0 + 32);
    }
    v16bf bg = glb_b_frag_f32(wg + k0);
    v16bf bu = glb_b_frag_f32(wu + k0);
#pragma unroll
    for (int m = 0; m < 8; ++m) {
      v16bf a = lds_a_frag(As[cur], m * 16 + r, half);
      accg[m] = WMMA_BF16(a, bg, accg[m]);
      accu[m] = WMMA_BF16(a, bu, accu[m]);
    }
    __syncthreads();
  }

  // epilogue: h = gelu(gate) * up, bf16 into [E, CAP, MID]
#pragma unroll
  for (int m = 0; m < 8; ++m) {
#pragma unroll
    for (int i = 0; i < 8; ++i) {
      int row = mbase + m * 16 + i + half * 8;
      if (row < C) {
        float hval = gelu_tanh(accg[m][i]) * accu[m][i];
        hbuf[((size_t)e * CAPd + row) * MIDd + jcol] = (__bf16)hval;
      }
    }
  }
}

// ---- kernel 4: GEMM2 + scaled atomic combine -------------------------------
// A tile (h, already bf16) staged via global_load_async_to_lds_b128 with
// double buffering; B = down weights streamed fp32->bf16.
// Grid: (H/128, CAP/128, E).

__global__ __launch_bounds__(256)
void moe_gemm2(const __bf16* __restrict__ hbuf, const float* __restrict__ down,
               const int* __restrict__ counts, const int* __restrict__ etok,
               const float* __restrict__ esc, float* __restrict__ out) {
  const int e = blockIdx.z;
  int C = counts[e]; if (C > CAPd) C = CAPd;
  const int mbase = blockIdx.y * 128;
  if (mbase >= C) return;
  const int nbase = blockIdx.x * 128;

  const int tid  = threadIdx.x;
  const int wave = tid >> 5, lane = tid & 31;
  const int half = lane >> 4, r = lane & 15;
  const int hcol = nbase + wave * 16 + r;  // 0..H-1

  __shared__ __align__(16) __bf16 As[2][128 * 32];

  const int arow = tid >> 1;
  const int akk  = (tid & 1) * 16;
  const bool avalid = (mbase + arow < C);

  // Async DMA addressing: uniform 64-bit base + per-lane 32-bit byte offset.
  const unsigned long long gbase =
      (unsigned long long)(uintptr_t)(hbuf + ((size_t)e * CAPd + mbase) * MIDd);
  // LDS byte address of this thread's slice in each buffer (low 32 bits of
  // the generic pointer = LDS aperture offset).
  const unsigned lds0 = (unsigned)(uintptr_t)&As[0][arow * 32 + akk];
  const unsigned lds1 = (unsigned)(uintptr_t)&As[1][arow * 32 + akk];

  auto stage_async = [&](int b, int k0) {
    if (!avalid) return;
    unsigned la = b ? lds1 : lds0;
    unsigned go = (unsigned)((arow * MIDd + akk + k0) * 2);  // bytes
    async_b128(la,      go,      gbase);
    async_b128(la + 16, go + 16, gbase);
  };

  // Pre-zero both buffers once: async DMA only overwrites valid rows, so
  // rows >= C stay zero for the whole K loop.
  {
    uint4* z = (uint4*)&As[0][0];
#pragma unroll
    for (int i = tid; i < 2 * 128 * 32 / 8; i += 256) z[i] = make_uint4(0u, 0u, 0u, 0u);
  }
  __syncthreads();

  const float* wd = down + ((size_t)e * Hdim + hcol) * MIDd + half * 16;

  v8f acc[8] = {};

  constexpr int NSTEP = MIDd / 32;  // 16 k-steps
  stage_async(0, 0);
  wait_async0();
  __syncthreads();

  for (int ks = 0; ks < NSTEP; ++ks) {
    const int cur = ks & 1;
    const int k0  = ks * 32;
    if (ks + 1 < NSTEP) {
      __builtin_prefetch(wd + k0 + 32, 0, 0);
      stage_async(cur ^ 1, k0 + 32);  // DMA next tile while computing
    }
    v16bf b = glb_b_frag_f32(wd + k0);
#pragma unroll
    for (int m = 0; m < 8; ++m) {
      v16bf a = lds_a_frag(As[cur], m * 16 + r, half);
      acc[m] = WMMA_BF16(a, b, acc[m]);
    }
    wait_async0();     // this wave's DMA for the next tile is in LDS
    __syncthreads();   // everyone's DMA done + all reads of 'cur' finished
  }

#pragma unroll
  for (int m = 0; m < 8; ++m) {
#pragma unroll
    for (int i = 0; i < 8; ++i) {
      int row = mbase + m * 16 + i + half * 8;
      if (row < C) {
        int   t = etok[e * CAPd + row];
        float s = esc [e * CAPd + row];
        atomicAdd(out + (size_t)t * Hdim + hcol, s * acc[m][i]);
      }
    }
  }
}

// ---- launcher --------------------------------------------------------------

extern "C" void kernel_launch(void* const* d_in, const int* in_sizes, int n_in,
                              void* d_out, int out_size, void* d_ws, size_t ws_size,
                              hipStream_t stream) {
  const float* x       = (const float*)d_in[0];
  const float* rscale  = (const float*)d_in[1];
  const float* rw      = (const float*)d_in[2];
  const float* pescale = (const float*)d_in[3];
  const float* gup     = (const float*)d_in[4];
  const float* down    = (const float*)d_in[5];
  // d_in[6] = capacity (device scalar); tiling statically assumes 512.

  const int T = in_sizes[0] / Hdim;

  // workspace carve-out
  char* ws = (char*)d_ws;
  int*    topk_idx = (int*)ws;                                     // T*K ints
  float*  topk_sc  = (float*)(ws + (size_t)T * KTOP * 4);          // T*K floats
  int*    counts   = (int*)(ws + (size_t)2 * T * KTOP * 4);        // E ints (pad 4KB)
  int*    etok     = (int*)(ws + (size_t)2 * T * KTOP * 4 + 4096); // E*CAP ints
  float*  esc      = (float*)((char*)etok + (size_t)Edim * CAPd * 4);
  __bf16* hbuf     = (__bf16*)((char*)esc + (size_t)Edim * CAPd * 4); // E*CAP*MID bf16 (64MB)

  float* out = (float*)d_out;
  const int n_out = T * Hdim;

  zero_init<<<dim3((n_out + 255) / 256), dim3(256), 0, stream>>>(out, n_out, counts);

  router_topk<<<dim3(T), dim3(128), 0, stream>>>(x, rscale, rw, pescale, topk_idx, topk_sc);

  assign_slots<<<dim3((T * KTOP + 255) / 256), dim3(256), 0, stream>>>(
      topk_idx, topk_sc, counts, etok, esc, T * KTOP);

  moe_gemm1<<<dim3(MIDd / 128, CAPd / 128, Edim), dim3(256), 0, stream>>>(
      x, gup, counts, etok, hbuf);

  moe_gemm2<<<dim3(Hdim / 128, CAPd / 128, Edim), dim3(256), 0, stream>>>(
      hbuf, down, counts, etok, esc, out);
}
